// SVDNet_62843961475343
// MI455X (gfx1250) — matrix operand order repaired and checked by reading, not compiled, not run on hardware
//
#include <hip/hip_runtime.h>

typedef __attribute__((ext_vector_type(16))) _Float16 v16h;
typedef __attribute__((ext_vector_type(8)))  float    v8f;

// ---------------------------------------------------------------- utilities
__global__ void f32_to_f16_kernel(const float* __restrict__ src,
                                  _Float16* __restrict__ dst, int n) {
    int i = blockIdx.x * blockDim.x + threadIdx.x;
    if (i < n) dst[i] = (_Float16)src[i];
}

// --------------------------------------------- weight swizzle for A fragments
// OIHW f32 -> wsz[mt][tap][cc][lane][t] f16, where (lane,t) is this code's
// contiguous-k 16x32 f16 A-fragment order: kk = t + 16*(lane>=16), m = mt*16+(lane&15)
template<int CIN, int COUT>
__global__ __launch_bounds__(256)
void wswizzle_kernel(const float* __restrict__ w, _Float16* __restrict__ wsz) {
    constexpr int CC = CIN / 32;
    constexpr int total = (COUT / 16) * 9 * CC * 32 * 16;
    int idx = blockIdx.x * 256 + threadIdx.x;
    if (idx >= total) return;
    int t    = idx & 15;
    int lane = (idx >> 4) & 31;
    int rest = idx >> 9;
    int cc   = rest % CC;  rest /= CC;
    int tap  = rest % 9;   int mt = rest / 9;
    int m    = mt * 16 + (lane & 15);
    int kk   = t + ((lane >= 16) ? 16 : 0);
    int cin  = cc * 32 + kk;
    wsz[idx] = (_Float16)w[(m * CIN + cin) * 9 + tap];
}

// ---------------------------------------------- LDS-staged tap-decomposed conv
// D(COUT x H*W) = sum_{tap} Wtap(COUT x CIN) * shift_tap(In)(CIN x H*W) + bias
// Block: 256 thr / 8 waves -> 16(M) x 128(N) tile, N-tile is 128 pixels of one
// image row (requires 128 | W).  Inner loop: A = 32B global load (pre-swizzled),
// B = 32B ds_load from channel-contiguous LDS tile, 1 WMMA.
template<int CIN, int COUT, int H, int W>
__global__ __launch_bounds__(256)
void conv3x3_wmma_lds(const _Float16* __restrict__ wsz,
                      const float*    __restrict__ bias,
                      const _Float16* __restrict__ in_h,   // CHW f16
                      _Float16*       __restrict__ out_h)  // CHW f16, relu'd
{
    constexpr int CC = CIN / 32;
    constexpr int HW = H * W;
    __shared__ alignas(32) _Float16 smem[3 * 132 * 32];   // [dy][x'(130,pad132)][cin]
    const int tid  = threadIdx.x;
    const int lane = tid & 31, wave = tid >> 5;
    const int hi   = lane >> 4, lo = lane & 15;
    const int nbase = blockIdx.x * 128;
    const int mbase = blockIdx.y * 16;
    const int ybase = nbase / W, xbase = nbase % W;        // whole tile on one row
    const int mt = mbase >> 4;
    const int ncol = nbase + wave * 16 + lo;
    const int xoffbase = wave * 16 + lo;                   // x'=0 <-> x = xbase-1

    v8f acc;
#pragma unroll
    for (int r = 0; r < 8; ++r) acc[r] = bias[mbase + r + 8 * hi];

    for (int cc = 0; cc < CC; ++cc) {
        __syncthreads();
        // ---- stage input chunk [3 rows][130 cols][32 ch] into LDS
        for (int idx = tid; idx < 3 * 130 * 32; idx += 256) {
            int xp   = idx % 130;
            int rest = idx / 130;
            int cin  = rest & 31;
            int dyi  = rest >> 5;
            int y = ybase + dyi - 1;
            int x = xbase + xp - 1;
            _Float16 v = (_Float16)0.f;
            if (y >= 0 && y < H && x >= 0 && x < W)
                v = in_h[(cc * 32 + cin) * HW + y * W + x];
            smem[(dyi * 132 + xp) * 32 + cin] = v;
        }
        __syncthreads();
        // ---- prefetch a slice of the next chunk (global_prefetch_b8)
        if (cc + 1 < CC) {
            int xp = tid % 130, rest = tid / 130;
            int cin = rest & 31, dyi = rest >> 5;
            int y = ybase + dyi - 1, x = xbase + xp - 1;
            if (y >= 0 && y < H && x >= 0 && x < W)
                __builtin_prefetch(in_h + ((cc + 1) * 32 + cin) * HW + y * W + x, 0, 1);
        }
        // ---- 9 taps, one WMMA each
#pragma unroll
        for (int tap = 0; tap < 9; ++tap) {
            const int dyi = tap / 3, dx = tap % 3;
            v16h va = *(const v16h*)(wsz + ((((size_t)mt * 9 + tap) * CC + cc) * 32 + lane) * 16);
            v16h vb = *(const v16h*)(&smem[(dyi * 132 + xoffbase + dx) * 32 + hi * 16]);
            acc = __builtin_amdgcn_wmma_f32_16x16x32_f16(
                      false, va, false, vb, (short)0, acc, false, false);
        }
    }
#pragma unroll
    for (int r = 0; r < 8; ++r) {
        int m = mbase + r + 8 * hi;
        float v = acc[r];
        out_h[m * HW + ncol] = (_Float16)(v > 0.f ? v : 0.f);
    }
}

// ------------------------------------------ conv1 (tiny K): gather-style WMMA
template<int CIN, int COUT, int H, int W>
__global__ __launch_bounds__(256)
void conv3x3_wmma_hwc(const _Float16* __restrict__ wh,   // [COUT][CIN*9] f16
                      const float*    __restrict__ bias,
                      const float*    __restrict__ in_f, // HWC f32, C=CIN
                      _Float16*       __restrict__ out_h)
{
    constexpr int K  = CIN * 9;
    constexpr int KC = (K + 31) / 32;
    constexpr int N  = H * W;
    const int lane = threadIdx.x & 31, wave = threadIdx.x >> 5;
    const int hi = lane >> 4, lo = lane & 15;
    const int nbase = (blockIdx.x * 8 + wave) * 16;
    const int mbase = blockIdx.y * 16;

    v8f acc;
#pragma unroll
    for (int r = 0; r < 8; ++r) acc[r] = bias[mbase + r + 8 * hi];

    const int mrow = mbase + lo;
    const int ncol = nbase + lo;
    const int py = ncol / W, px = ncol % W;

    for (int kc = 0; kc < KC; ++kc) {
        v16h va = {}; v16h vb = {};
#pragma unroll
        for (int t = 0; t < 16; ++t) {
            int kA = kc * 32 + t + (hi ? 16 : 0);
            va[t] = (kA < K) ? wh[mrow * K + kA] : (_Float16)0.f;
            int kB = kc * 32 + t + (hi ? 16 : 0);
            _Float16 bv = (_Float16)0.f;
            if (kB < K) {
                int cin = kB / 9;
                int rem = kB - cin * 9;
                int dy = rem / 3 - 1;
                int dx = rem - (rem / 3) * 3 - 1;
                int yy = py + dy, xx = px + dx;
                if (yy >= 0 && yy < H && xx >= 0 && xx < W)
                    bv = (_Float16)in_f[(yy * W + xx) * CIN + cin];
            }
            vb[t] = bv;
        }
        acc = __builtin_amdgcn_wmma_f32_16x16x32_f16(
                  false, va, false, vb, (short)0, acc, false, false);
    }
#pragma unroll
    for (int r = 0; r < 8; ++r) {
        int m = mbase + r + 8 * hi;
        float v = acc[r];
        out_h[m * N + ncol] = (_Float16)(v > 0.f ? v : 0.f);
    }
}

// ------------------------------------------------------------------ maxpool
__global__ void maxpool2_kernel(const _Float16* __restrict__ in,
                                _Float16* __restrict__ out) {
    int idx = blockIdx.x * blockDim.x + threadIdx.x;   // 128*128*128 outputs
    if (idx >= 128 * 128 * 128) return;
    int c = idx >> 14, r = idx & 16383;
    int oy = r >> 7, ox = r & 127;
    const _Float16* p = in + c * 65536 + (oy * 2) * 256 + ox * 2;
    float a = (float)p[0], b = (float)p[1], cc = (float)p[256], d = (float)p[257];
    out[idx] = (_Float16)fmaxf(fmaxf(a, b), fmaxf(cc, d));
}

// ------------------------------------------------------- global average pool
__global__ __launch_bounds__(256)
void chan_mean_kernel(const _Float16* __restrict__ act, float* __restrict__ feat) {
    __shared__ float red[256];
    int c = blockIdx.x;
    float s = 0.f;
    for (int i = threadIdx.x; i < 16384; i += 256) s += (float)act[c * 16384 + i];
    red[threadIdx.x] = s; __syncthreads();
    for (int st = 128; st > 0; st >>= 1) {
        if (threadIdx.x < st) red[threadIdx.x] += red[threadIdx.x + st];
        __syncthreads();
    }
    if (threadIdx.x == 0) feat[c] = red[0] * (1.0f / 16384.0f);
}

// --------------------------------------------------------------- small GEMV
__global__ __launch_bounds__(256)
void gemv_kernel(const float* __restrict__ x, const float* __restrict__ W,
                 const float* __restrict__ b, float* __restrict__ y,
                 int K, int N, int relu) {
    __shared__ float xs[1024];
    for (int i = threadIdx.x; i < K; i += blockDim.x) xs[i] = x[i];
    __syncthreads();
    int n = blockIdx.x * blockDim.x + threadIdx.x;
    if (n >= N) return;
    float acc = b[n];
    for (int k = 0; k < K; ++k) acc = fmaf(xs[k], W[(size_t)k * N + n], acc);
    y[n] = relu ? fmaxf(acc, 0.f) : acc;
}

// --------------------------------------- bandwidth GEMV (128 MB weight stream)
__global__ __launch_bounds__(256)
void gemv4_kernel(const float* __restrict__ x, const float4* __restrict__ W4,
                  const float4* __restrict__ b4, float* __restrict__ y,
                  int K, int N4) {
    __shared__ float xs[1024];
    for (int i = threadIdx.x; i < K; i += 256) xs[i] = x[i];
    __syncthreads();
    int n = blockIdx.x * 256 + threadIdx.x;
    if (n >= N4) return;
    float4 acc = b4[n];
    for (int k = 0; k < K; ++k) {
        float4 w = W4[(size_t)k * N4 + n];
        float xv = xs[k];
        acc.x = fmaf(xv, w.x, acc.x); acc.y = fmaf(xv, w.y, acc.y);
        acc.z = fmaf(xv, w.z, acc.z); acc.w = fmaf(xv, w.w, acc.w);
    }
    ((float4*)y)[n] = acc;
}

// ------------------------------------------------- softplus + descending sort
__global__ void softplus_sort_kernel(const float* __restrict__ spre,
                                     float* __restrict__ out) {
    __shared__ float sv[64];
    int t = threadIdx.x;
    float x = spre[t];
    sv[t] = (x > 20.f) ? x : log1pf(expf(x));
    __syncthreads();
    float mine = sv[t];
    int rank = 0;
    for (int j = 0; j < 64; ++j) {
        float o = sv[j];
        if (o > mine || (o == mine && j < t)) rank++;
    }
    out[rank] = mine;
}

// -------------------------------------------- complex modified Gram-Schmidt
__global__ __launch_bounds__(256)
void mgs_kernel(const float* __restrict__ upre, const float* __restrict__ vpre,
                float* __restrict__ Qr_u, float* __restrict__ Qi_u,
                float* __restrict__ Qr_v, float* __restrict__ Qi_v,
                float* __restrict__ out_u, float* __restrict__ out_v) {
    const float* pre; float* Qr; float* Qi; float* out;
    if (blockIdx.x == 0) { pre = upre; Qr = Qr_u; Qi = Qi_u; out = out_u; }
    else                 { pre = vpre; Qr = Qr_v; Qi = Qi_v; out = out_v; }
    __shared__ float rr[256], ri[256];
    const int t = threadIdx.x, R = 64, M = 256;
    for (int j = 0; j < R; ++j) {
        float vr = pre[t * R * 2 + j * 2 + 0];
        float vi = pre[t * R * 2 + j * 2 + 1];
        for (int i = 0; i < j; ++i) {
            float qr = Qr[i * M + t], qi = Qi[i * M + t];
            rr[t] = qr * vr + qi * vi;
            ri[t] = qr * vi - qi * vr;
            __syncthreads();
            for (int st = 128; st > 0; st >>= 1) {
                if (t < st) { rr[t] += rr[t + st]; ri[t] += ri[t + st]; }
                __syncthreads();
            }
            float dr = rr[0], di = ri[0];
            __syncthreads();
            vr -= dr * qr - di * qi;
            vi -= dr * qi + di * qr;
        }
        rr[t] = vr * vr + vi * vi;
        __syncthreads();
        for (int st = 128; st > 0; st >>= 1) {
            if (t < st) rr[t] += rr[t + st];
            __syncthreads();
        }
        float inv = 1.0f / sqrtf(rr[0] + 1e-8f);
        __syncthreads();
        float qr = vr * inv, qi = vi * inv;
        Qr[j * M + t] = qr; Qi[j * M + t] = qi;
        out[t * R * 2 + j * 2 + 0] = qr;
        out[t * R * 2 + j * 2 + 1] = qi;
        __syncthreads();
    }
}

// ============================================================== launch glue
extern "C" void kernel_launch(void* const* d_in, const int* in_sizes, int n_in,
                              void* d_out, int out_size, void* d_ws, size_t ws_size,
                              hipStream_t stream) {
    const float* x   = (const float*)d_in[0];
    const float* cw1 = (const float*)d_in[1];  const float* cb1 = (const float*)d_in[2];
    const float* cw2 = (const float*)d_in[3];  const float* cb2 = (const float*)d_in[4];
    const float* cw3 = (const float*)d_in[5];  const float* cb3 = (const float*)d_in[6];
    const float* lw  = (const float*)d_in[7];  const float* lb  = (const float*)d_in[8];
    const float* sw1 = (const float*)d_in[9];  const float* sb1 = (const float*)d_in[10];
    const float* sw2 = (const float*)d_in[11]; const float* sb2 = (const float*)d_in[12];
    const float* uw1 = (const float*)d_in[13]; const float* ub1 = (const float*)d_in[14];
    const float* uw2 = (const float*)d_in[15]; const float* ub2 = (const float*)d_in[16];
    const float* vw1 = (const float*)d_in[17]; const float* vb1 = (const float*)d_in[18];
    const float* vw2 = (const float*)d_in[19]; const float* vb2 = (const float*)d_in[20];
    float* out = (float*)d_out;                    // [u(32768) | s(64) | v(32768)]

    // ------- workspace carve-up
    size_t off = 0;
    auto carve = [&](size_t bytes) -> void* {
        void* p = (char*)d_ws + off;
        off = (off + bytes + 255) & ~(size_t)255;
        return p;
    };
    _Float16* w1h   = (_Float16*)carve(64  * 18   * 2);
    _Float16* w2sz  = (_Float16*)carve(128 * 576  * 2);   // swizzled
    _Float16* w3sz  = (_Float16*)carve(256 * 1152 * 2);   // swizzled
    _Float16* act1  = (_Float16*)carve((size_t)64  * 65536 * 2);
    _Float16* act2  = (_Float16*)carve((size_t)128 * 65536 * 2);
    _Float16* act2p = (_Float16*)carve((size_t)128 * 16384 * 2);
    _Float16* act3  = (_Float16*)carve((size_t)256 * 16384 * 2);
    float* feat  = (float*)carve(256  * 4);
    float* feat2 = (float*)carve(512  * 4);
    float* hs    = (float*)carve(256  * 4);
    float* spre  = (float*)carve(64   * 4);
    float* hu    = (float*)carve(1024 * 4);
    float* hv    = (float*)carve(1024 * 4);
    float* upre  = (float*)carve(32768 * 4);
    float* vpre  = (float*)carve(32768 * 4);
    float* Qr_u  = (float*)carve(64 * 256 * 4);
    float* Qi_u  = (float*)carve(64 * 256 * 4);
    float* Qr_v  = (float*)carve(64 * 256 * 4);
    float* Qi_v  = (float*)carve(64 * 256 * 4);

    // ------- weight prep
    f32_to_f16_kernel<<<(64*18 + 255)/256, 256, 0, stream>>>(cw1, w1h, 64*18);
    wswizzle_kernel<64, 128><<<(128*576  + 255)/256, 256, 0, stream>>>(cw2, w2sz);
    wswizzle_kernel<128,256><<<(256*1152 + 255)/256, 256, 0, stream>>>(cw3, w3sz);

    // ------- conv stack
    conv3x3_wmma_hwc<2, 64, 256, 256><<<dim3(512, 4), 256, 0, stream>>>(w1h, cb1, x, act1);
    conv3x3_wmma_lds<64, 128, 256, 256><<<dim3(512, 8),  256, 0, stream>>>(w2sz, cb2, act1, act2);
    maxpool2_kernel<<<(128*128*128 + 255)/256, 256, 0, stream>>>(act2, act2p);
    conv3x3_wmma_lds<128, 256, 128, 128><<<dim3(128, 16), 256, 0, stream>>>(w3sz, cb3, act2p, act3);
    chan_mean_kernel<<<256, 256, 0, stream>>>(act3, feat);

    // ------- dense trunk + heads
    gemv_kernel<<<2, 256, 0, stream>>>(feat,  lw,  lb,  feat2, 256, 512,  1);
    gemv_kernel<<<1, 256, 0, stream>>>(feat2, sw1, sb1, hs,    512, 256,  1);
    gemv_kernel<<<1, 256, 0, stream>>>(hs,    sw2, sb2, spre,  256, 64,   0);
    gemv_kernel<<<4, 256, 0, stream>>>(feat2, uw1, ub1, hu,    512, 1024, 1);
    gemv_kernel<<<4, 256, 0, stream>>>(feat2, vw1, vb1, hv,    512, 1024, 1);
    gemv4_kernel<<<32, 256, 0, stream>>>(hu, (const float4*)uw2, (const float4*)ub2, upre, 1024, 8192);
    gemv4_kernel<<<32, 256, 0, stream>>>(hv, (const float4*)vw2, (const float4*)vb2, vpre, 1024, 8192);

    // ------- s: softplus + descending sort
    softplus_sort_kernel<<<1, 64, 0, stream>>>(spre, out + 32768);

    // ------- complex MGS for U and V
    mgs_kernel<<<2, 256, 0, stream>>>(upre, vpre, Qr_u, Qi_u, Qr_v, Qi_v,
                                      out, out + 32768 + 64);
}
